// TransNet_81217831567648
// MI455X (gfx1250) — matrix operand order repaired and checked by reading, not compiled, not run on hardware
//
#include <hip/hip_runtime.h>
#include <hip/hip_bf16.h>
#include <math.h>

typedef _Float16 v16h __attribute__((ext_vector_type(16)));
typedef float    v8f  __attribute__((ext_vector_type(8)));

#define TILE_M 128
#define TILE_N 128
#define TILE_K 32
#define LDS_STRIDE 56   // halves: 112B rows -> 16B aligned, conflict-free b128 row reads

union Frag {
  v16h v;
  uint4 q[2];
};

__device__ __forceinline__ v8f wmma3(const Frag& ah, const Frag& al,
                                     const Frag& bh, const Frag& bl, v8f c) {
  // split-f16: exact ~ Ahi*Bhi + Ahi*Blo + Alo*Bhi  (lo*lo dropped, ~2e-7 rel)
  c = __builtin_amdgcn_wmma_f32_16x16x32_f16(false, ah.v, false, bh.v, (short)0, c, false, false);
  c = __builtin_amdgcn_wmma_f32_16x16x32_f16(false, ah.v, false, bl.v, (short)0, c, false, false);
  c = __builtin_amdgcn_wmma_f32_16x16x32_f16(false, al.v, false, bh.v, (short)0, c, false, false);
  return c;
}

// C[M,N] = alpha * A(fp32)[M,K] @ Bt(f16 hi/lo)[N,K]^T  (+ bias[n]) (+ beta*D) (+ diag*I)
// optionally C2 = relu(C)
__global__ __launch_bounds__(256)
void gemm_f16split(const float* __restrict__ A, int lda,
                   const _Float16* __restrict__ Bh, const _Float16* __restrict__ Bl, int ldb,
                   float* __restrict__ C, int ldc,
                   float* __restrict__ C2,
                   const float* __restrict__ Dm,
                   const float* __restrict__ bias,
                   int M, int N, int K,
                   float alpha, float beta, float diagv)
{
  __shared__ _Float16 AsH[TILE_M * LDS_STRIDE];
  __shared__ _Float16 AsL[TILE_M * LDS_STRIDE];
  __shared__ _Float16 BsH[TILE_N * LDS_STRIDE];
  __shared__ _Float16 BsL[TILE_N * LDS_STRIDE];

  const int t   = threadIdx.x;
  const int bm0 = blockIdx.y * TILE_M;
  const int bn0 = blockIdx.x * TILE_N;

  const int w    = t >> 5;      // wave id 0..7
  const int lane = t & 31;
  const int wm   = w & 1;       // 2 waves along M -> 64 rows each
  const int wn   = w >> 1;      // 4 waves along N -> 32 cols each
  const int r    = lane & 15;
  const int hi16 = lane >> 4;

  const v8f vzero = {0.f,0.f,0.f,0.f,0.f,0.f,0.f,0.f};
  v8f acc[4][2];
#pragma unroll
  for (int i = 0; i < 4; i++)
#pragma unroll
    for (int j = 0; j < 2; j++) acc[i][j] = vzero;

  const int arow = t >> 1;          // 0..127
  const int acol = (t & 1) * 16;    // 0 or 16

  for (int k0 = 0; k0 < K; k0 += TILE_K) {
    // ---- stage A tile: fp32 global -> (hi,lo) f16 in LDS ----
    {
      const float4* ap4 = (const float4*)(A + (size_t)(bm0 + arow) * lda + k0 + acol);
#pragma unroll
      for (int j = 0; j < 4; j++) {
        float4 f = ap4[j];
        float xs[4] = {f.x, f.y, f.z, f.w};
        union Pack { _Float16 h[4]; uint2 u2; } ph, pl;
#pragma unroll
        for (int e = 0; e < 4; e++) {
          _Float16 h = (_Float16)xs[e];
          ph.h[e] = h;
          pl.h[e] = (_Float16)(xs[e] - (float)h);
        }
        int bidx = arow * LDS_STRIDE + acol + j * 4;
        *(uint2*)(&AsH[bidx]) = ph.u2;
        *(uint2*)(&AsL[bidx]) = pl.u2;
      }
    }
    // ---- stage B tile: pre-split f16 global -> LDS (zero-fill past N) ----
#pragma unroll
    for (int i = 0; i < 2; i++) {
      int idx   = t * 2 + i;
      int brow  = idx >> 2;
      int chunk = idx & 3;
      int gn    = bn0 + brow;
      uint4 qh = {0u,0u,0u,0u}, ql = {0u,0u,0u,0u};
      if (gn < N) {
        const uint4* bph = (const uint4*)(Bh + (size_t)gn * ldb + k0);
        const uint4* bpl = (const uint4*)(Bl + (size_t)gn * ldb + k0);
        qh = bph[chunk];
        ql = bpl[chunk];
      }
      *(uint4*)(&BsH[brow * LDS_STRIDE + chunk * 8]) = qh;
      *(uint4*)(&BsL[brow * LDS_STRIDE + chunk * 8]) = ql;
    }
    __syncthreads();

    // ---- build fragments per ISA 16-bit layouts ----
    Frag aH[4], aL[4], bH[2], bL[2];
    const int ka = hi16 * 8;    // A: lanes16-31 start at K=8 (and K=24)
    const int kb = hi16 * 8;    // B: lanes16-31 start at K=8 (and K=24)
#pragma unroll
    for (int mt = 0; mt < 4; mt++) {
      int row = wm * 64 + mt * 16 + r;
      const _Float16* ph = &AsH[row * LDS_STRIDE];
      const _Float16* pl = &AsL[row * LDS_STRIDE];
      aH[mt].q[0] = *(const uint4*)(ph + ka);
      aH[mt].q[1] = *(const uint4*)(ph + ka + 16);
      aL[mt].q[0] = *(const uint4*)(pl + ka);
      aL[mt].q[1] = *(const uint4*)(pl + ka + 16);
    }
#pragma unroll
    for (int nt = 0; nt < 2; nt++) {
      int row = wn * 32 + nt * 16 + r;
      const _Float16* ph = &BsH[row * LDS_STRIDE];
      const _Float16* pl = &BsL[row * LDS_STRIDE];
      bH[nt].q[0] = *(const uint4*)(ph + kb);
      bH[nt].q[1] = *(const uint4*)(ph + kb + 16);
      bL[nt].q[0] = *(const uint4*)(pl + kb);
      bL[nt].q[1] = *(const uint4*)(pl + kb + 16);
    }
#pragma unroll
    for (int mt = 0; mt < 4; mt++)
#pragma unroll
      for (int nt = 0; nt < 2; nt++)
        acc[mt][nt] = wmma3(aH[mt], aL[mt], bH[nt], bL[nt], acc[mt][nt]);
    __syncthreads();
  }

  // ---- epilogue (C VGPR layout: vgpr g -> M=g or g+8; lane&15 -> N) ----
#pragma unroll
  for (int mt = 0; mt < 4; mt++) {
#pragma unroll
    for (int nt = 0; nt < 2; nt++) {
      int n = bn0 + wn * 32 + nt * 16 + r;
      bool nok = n < N;
      float bv = (bias != nullptr && nok) ? bias[n] : 0.0f;
#pragma unroll
      for (int g = 0; g < 8; g++) {
        int m = bm0 + wm * 64 + mt * 16 + hi16 * 8 + g;
        float val = alpha * acc[mt][nt][g] + bv;
        if (Dm != nullptr && nok) val += beta * Dm[(size_t)m * ldc + n];
        if (diagv != 0.0f && m == n) val += diagv;
        if (nok) {
          C[(size_t)m * ldc + n] = val;
          if (C2 != nullptr) C2[(size_t)m * ldc + n] = fmaxf(val, 0.0f);
        }
      }
    }
  }
}

// rhs_u = u + DT*b[col]; rhs_v = v + (DT/EPS)*relu(u)   (in place)
__global__ void rhs_uv(float* __restrict__ Ub, float* __restrict__ Vb,
                       const float* __restrict__ b, int colmask, size_t n)
{
  size_t i = (size_t)blockIdx.x * blockDim.x + threadIdx.x;
  if (i >= n) return;
  float u = Ub[i], v = Vb[i];
  int col = (int)(i & (size_t)colmask);
  Ub[i] = u + 0.1f * b[col];
  Vb[i] = v + 10.0f * fmaxf(u, 0.0f);
}

__global__ void transpose_f32(const float* __restrict__ in, float* __restrict__ out,
                              int R, int C)
{
  __shared__ float tile[16][17];
  int bx = blockIdx.x * 16, by = blockIdx.y * 16;
  tile[threadIdx.y][threadIdx.x] = in[(size_t)(by + threadIdx.y) * C + bx + threadIdx.x];
  __syncthreads();
  out[(size_t)(bx + threadIdx.y) * R + by + threadIdx.x] = tile[threadIdx.x][threadIdx.y];
}

__global__ void split_hi_lo(const float* __restrict__ in,
                            _Float16* __restrict__ hi, _Float16* __restrict__ lo, size_t n)
{
  size_t i = (size_t)blockIdx.x * blockDim.x + threadIdx.x;
  if (i >= n) return;
  float x = in[i];
  _Float16 h = (_Float16)x;
  hi[i] = h;
  lo[i] = (_Float16)(x - (float)h);
}

__global__ __launch_bounds__(256)
void softmax_rows(const float* __restrict__ in, float* __restrict__ out, int cols)
{
  __shared__ float red[256];
  const int row = blockIdx.x;
  const float* x = in + (size_t)row * cols;
  float lmax = -3.4e38f;
  for (int c = threadIdx.x; c < cols; c += 256) lmax = fmaxf(lmax, x[c]);
  red[threadIdx.x] = lmax; __syncthreads();
  for (int s = 128; s > 0; s >>= 1) {
    if (threadIdx.x < s) red[threadIdx.x] = fmaxf(red[threadIdx.x], red[threadIdx.x + s]);
    __syncthreads();
  }
  const float m = red[0];
  __syncthreads();
  float lsum = 0.f;
  for (int c = threadIdx.x; c < cols; c += 256) lsum += expf(x[c] - m);
  red[threadIdx.x] = lsum; __syncthreads();
  for (int s = 128; s > 0; s >>= 1) {
    if (threadIdx.x < s) red[threadIdx.x] += red[threadIdx.x + s];
    __syncthreads();
  }
  const float inv = 1.0f / red[0];
  for (int c = threadIdx.x; c < cols; c += 256)
    out[(size_t)row * cols + c] = expf(x[c] - m) * inv;
}

extern "C" void kernel_launch(void* const* d_in, const int* in_sizes, int n_in,
                              void* d_out, int out_size, void* d_ws, size_t ws_size,
                              hipStream_t stream) {
  const float* x     = (const float*)d_in[0];
  const float* W_in  = (const float*)d_in[1];
  const float* b_in  = (const float*)d_in[2];
  const float* Wl[3] = {(const float*)d_in[3], (const float*)d_in[5], (const float*)d_in[7]};
  const float* bl[3] = {(const float*)d_in[4], (const float*)d_in[6], (const float*)d_in[8]};
  const float* W_out = (const float*)d_in[9];
  const float* b_out = (const float*)d_in[10];
  float* out = (float*)d_out;

  const int B = 8192, U = 1024, IN = 2048, OUT = 1000;
  const float DT = 0.1f;
  const float RC = 11.0f;               // c = 1 + DT/EPS

  char* base = (char*)d_ws;
  size_t off = 0;
  auto alloc = [&](size_t bytes) -> void* {
    void* p = base + off;
    off += (bytes + 255) & ~(size_t)255;
    return p;
  };

  // activation ping-pong buffers (fp32)
  float* P0 = (float*)alloc((size_t)B * U * 4);
  float* P1 = (float*)alloc((size_t)B * U * 4);
  float* P2 = (float*)alloc((size_t)B * U * 4);
  // split-f16 weights
  _Float16* WinH  = (_Float16*)alloc((size_t)U * IN * 2);
  _Float16* WinL  = (_Float16*)alloc((size_t)U * IN * 2);
  _Float16* WoutH = (_Float16*)alloc((size_t)OUT * U * 2);
  _Float16* WoutL = (_Float16*)alloc((size_t)OUT * U * 2);
  _Float16 *WH[3], *WL[3], *WTH[3], *WTL[3], *MH[3], *ML[3];
  for (int i = 0; i < 3; i++) {
    WH[i]  = (_Float16*)alloc((size_t)U * U * 2);
    WL[i]  = (_Float16*)alloc((size_t)U * U * 2);
    WTH[i] = (_Float16*)alloc((size_t)U * U * 2);
    WTL[i] = (_Float16*)alloc((size_t)U * U * 2);
    MH[i]  = (_Float16*)alloc((size_t)U * U * 2);
    ML[i]  = (_Float16*)alloc((size_t)U * U * 2);
  }
  // fp32 temporaries (reused per layer)
  float* WTf = (float*)alloc((size_t)U * U * 4);
  float* Ef  = (float*)alloc((size_t)U * U * 4);
  float* Mf  = (float*)alloc((size_t)U * U * 4);
  _Float16* EH = (_Float16*)alloc((size_t)U * U * 2);
  _Float16* EL = (_Float16*)alloc((size_t)U * U * 2);

  auto gemm = [&](const float* A, int lda, const _Float16* Bh, const _Float16* Bl, int ldb,
                  float* C, int ldc, float* C2, const float* Dm, const float* bias,
                  int M, int N, int K, float alpha, float beta, float diagv) {
    dim3 g((unsigned)((N + TILE_N - 1) / TILE_N), (unsigned)(M / TILE_M));
    gemm_f16split<<<g, 256, 0, stream>>>(A, lda, Bh, Bl, ldb, C, ldc, C2, Dm, bias,
                                         M, N, K, alpha, beta, diagv);
  };
  auto split = [&](const float* in, _Float16* hi, _Float16* lo, size_t n) {
    split_hi_lo<<<(unsigned)((n + 255) / 256), 256, 0, stream>>>(in, hi, lo, n);
  };

  // ---------------- weight precompute ----------------
  split(W_in, WinH, WinL, (size_t)U * IN);
  split(W_out, WoutH, WoutL, (size_t)OUT * U);
  for (int i = 0; i < 3; i++) {
    transpose_f32<<<dim3(U / 16, U / 16), dim3(16, 16), 0, stream>>>(Wl[i], WTf, U, U);
    split(Wl[i], WH[i], WL[i], (size_t)U * U);
    split(WTf, WTH[i], WTL[i], (size_t)U * U);
    // E = (DT^2/c) * W^T W  = (DT^2/c) * WT @ WT^T
    gemm(WTf, U, WTH[i], WTL[i], U, Ef, U, nullptr, nullptr, nullptr,
         U, U, U, DT * DT / RC, 0.f, 0.f);
    split(Ef, EH, EL, (size_t)U * U);
    // Minv = (1/c)(I - E + E@E)   (||E|| ~ 1.2e-3 -> error ~1e-9, below fp32 eps)
    gemm(Ef, U, EH, EL, U, Mf, U, nullptr, Ef, nullptr,
         U, U, U, 1.0f / RC, -1.0f / RC, 1.0f / RC);
    split(Mf, MH[i], ML[i], (size_t)U * U);
  }

  // ---------------- batch pipeline ----------------
  // z = x @ W_in^T + b_in ; u = z (P0), v = relu(z) (P1)
  gemm(x, IN, WinH, WinL, IN, P0, U, P1, nullptr, b_in, B, U, IN, 1.0f, 0.f, 0.f);

  float *Ub = P0, *Vb = P1, *Fb = P2;
  const size_t nact = (size_t)B * U;
  for (int i = 0; i < 3; i++) {
    // rhs_u = u + DT*b (in place in Ub); rhs_v = v + (DT/EPS)*relu(u) (in place in Vb)
    rhs_uv<<<(unsigned)((nact + 255) / 256), 256, 0, stream>>>(Ub, Vb, bl[i], U - 1, nact);
    // T = rhs_v + DT * rhs_u @ W        -> Vb
    gemm(Ub, U, WTH[i], WTL[i], U, Vb, U, nullptr, Vb, nullptr, B, U, U, DT, 1.0f, 0.f);
    // Vout = -v = -(T @ Minv)           -> Fb
    gemm(Vb, U, MH[i], ML[i], U, Fb, U, nullptr, nullptr, nullptr, B, U, U, -1.0f, 0.f, 0.f);
    // Uout = -u = -DT*(Vout @ W^T) - rhs_u   (v@W^T == -Vout@W^T)  -> Ub
    gemm(Fb, U, WH[i], WL[i], U, Ub, U, nullptr, Ub, nullptr, B, U, U, -DT, -1.0f, 0.f);
    // new state: U=Ub, V=Fb ; free = old Vb
    float* tmp = Vb; Vb = Fb; Fb = tmp;
  }

  // logits = u @ W_out^T + b_out  -> Fb (free buffer), then row softmax -> d_out
  gemm(Ub, U, WoutH, WoutL, U, Fb, OUT, nullptr, nullptr, b_out, B, OUT, U, 1.0f, 0.f, 0.f);
  softmax_rows<<<B, 256, 0, stream>>>(Fb, out, OUT);
}
